// DeepSeekV3MoE_38955353375116
// MI455X (gfx1250) — matrix operand order repaired and checked by reading, not compiled, not run on hardware
//
#include <hip/hip_runtime.h>

#define DH 2048   // hidden
#define II 1024   // intermediate
#define NE 16     // routed experts
#define NS 2      // shared experts
#define TT 8192   // tokens (B*SEQ)
#define MT 32     // tokens per block tile

typedef __attribute__((ext_vector_type(16))) __bf16          v16bf;
typedef __attribute__((ext_vector_type(8)))  float           v8f;

struct B32x2 { uint4 a, b; };

__device__ __forceinline__ unsigned short bfbits(float f) {
  __bf16 h = (__bf16)f;                       // native v_cvt to bf16
  return __builtin_bit_cast(unsigned short, h);
}

__device__ __forceinline__ float fast_rcp(float x) {
#if __has_builtin(__builtin_amdgcn_rcpf)
  return __builtin_amdgcn_rcpf(x);            // v_rcp_f32
#else
  return 1.0f / x;
#endif
}

__device__ __forceinline__ void wait_asynccnt0() {
#if __has_builtin(__builtin_amdgcn_s_wait_asynccnt)
  __builtin_amdgcn_s_wait_asynccnt(0);
#else
  asm volatile("s_wait_asynccnt 0x0" ::: "memory");
#endif
}

// A-fragment (16x32 bf16, M=lane&15): lane-half selects a block of 16 contiguous K values (K=kb+koff..kb+koff+15).
__device__ __forceinline__ v16bf frag_a_lds(const unsigned short* base, int ld, int kb, int lane) {
  const int M    = lane & 15;
  const int koff = (lane >> 4) << 4;
  B32x2 r;
  r.a = *(const uint4*)(base + (size_t)M * ld + kb + koff);
  r.b = *(const uint4*)(base + (size_t)M * ld + kb + 8 + koff);
  return __builtin_bit_cast(v16bf, r);
}

// B-fragment (32x16 bf16, N=lane&15): lane-half selects a block of 16 contiguous K values.
__device__ __forceinline__ v16bf frag_b(const unsigned short* row, int kb, int lane) {
  const unsigned short* p = row + kb + ((lane >> 4) << 4);
  B32x2 r;
  r.a = *(const uint4*)(p);
  r.b = *(const uint4*)(p + 8);
  return __builtin_bit_cast(v16bf, r);
}

__device__ __forceinline__ v16bf frag_b(const float* row, int kb, int lane) {
  const float* p = row + kb + ((lane >> 4) << 4);
  float4 f0 = *(const float4*)(p);
  float4 f1 = *(const float4*)(p + 4);
  float4 f2 = *(const float4*)(p + 8);
  float4 f3 = *(const float4*)(p + 12);
  v16bf u;
  u[0]  = (__bf16)f0.x; u[1]  = (__bf16)f0.y; u[2]  = (__bf16)f0.z; u[3]  = (__bf16)f0.w;
  u[4]  = (__bf16)f1.x; u[5]  = (__bf16)f1.y; u[6]  = (__bf16)f1.z; u[7]  = (__bf16)f1.w;
  u[8]  = (__bf16)f2.x; u[9]  = (__bf16)f2.y; u[10] = (__bf16)f2.z; u[11] = (__bf16)f2.w;
  u[12] = (__bf16)f3.x; u[13] = (__bf16)f3.y; u[14] = (__bf16)f3.z; u[15] = (__bf16)f3.w;
  return u;
}

// ---------------- helpers ----------------
__global__ void moe_zero(float* p, int n) {
  for (int i = blockIdx.x * blockDim.x + threadIdx.x; i < n; i += gridDim.x * blockDim.x)
    p[i] = 0.0f;
}

// f32 -> bf16 (packed pair writes)
__global__ void moe_cvt_bf16(const float* __restrict__ src, unsigned short* __restrict__ dst, int n) {
  const int stride = gridDim.x * blockDim.x * 2;
  for (int i = (blockIdx.x * blockDim.x + threadIdx.x) * 2; i < n; i += stride) {
    float2 f = *(const float2*)(src + i);
    *(unsigned*)(dst + i) = (unsigned)bfbits(f.x) | ((unsigned)bfbits(f.y) << 16);
  }
}

// ---------------- gating kernel (one wave32 per token) ----------------
__global__ __launch_bounds__(256)
void moe_gate(const float* __restrict__ x, const float* __restrict__ gw,
              const float* __restrict__ gb, float* __restrict__ comb,
              float* __restrict__ load) {
  const int t    = blockIdx.x * (blockDim.x >> 5) + (threadIdx.x >> 5);
  const int lane = threadIdx.x & 31;
  if (t >= TT) return;

  float acc[NE];
  #pragma unroll
  for (int e = 0; e < NE; ++e) acc[e] = 0.0f;

  const float* xp = x + (size_t)t * DH;
  for (int k = lane; k < DH; k += 32) {
    float xv = xp[k];
    #pragma unroll
    for (int e = 0; e < NE; ++e) acc[e] += xv * gw[e * DH + k];
  }
  #pragma unroll
  for (int e = 0; e < NE; ++e) {
    float v = acc[e];
    for (int off = 16; off; off >>= 1) v += __shfl_xor(v, off, 32);
    acc[e] = v;
  }

  if (lane == 0) {
    float sco[NE], s[NE];
    #pragma unroll
    for (int e = 0; e < NE; ++e) {
      sco[e] = fast_rcp(1.0f + __expf(-acc[e]));   // sigmoid (original scores)
      s[e]   = sco[e] + gb[e];
    }
    float gsc[4];
    #pragma unroll
    for (int g = 0; g < 4; ++g) {
      float m1 = -__builtin_inff(), m2 = -__builtin_inff();
      for (int i = 0; i < 4; ++i) {
        float v = s[g * 4 + i];
        if (v > m1) { m2 = m1; m1 = v; } else if (v > m2) { m2 = v; }
      }
      gsc[g] = m1 + m2;
    }
    int g0 = 0;
    for (int g = 1; g < 4; ++g) if (gsc[g] > gsc[g0]) g0 = g;
    int g1 = (g0 == 0) ? 1 : 0;
    for (int g = 0; g < 4; ++g) if (g != g0 && gsc[g] > gsc[g1]) g1 = g;
    float sm[NE];
    #pragma unroll
    for (int e = 0; e < NE; ++e) {
      int g = e >> 2;
      sm[e] = (g == g0 || g == g1) ? s[e] : -__builtin_inff();
    }
    int idx[4]; float wv[4]; float sum = 0.0f;
    #pragma unroll
    for (int j = 0; j < 4; ++j) {
      int best = 0;
      for (int e = 1; e < NE; ++e) if (sm[e] > sm[best]) best = e;
      idx[j] = best; wv[j] = sco[best]; sum += wv[j];
      sm[best] = -__builtin_inff();
    }
    float c[NE];
    #pragma unroll
    for (int e = 0; e < NE; ++e) c[e] = 0.0f;
    float rs = fast_rcp(sum) * 2.5f;
    #pragma unroll
    for (int j = 0; j < 4; ++j) {
      c[idx[j]] = wv[j] * rs;
      atomicAdd(&load[idx[j]], 1.0f);
    }
    #pragma unroll
    for (int e = 0; e < NE; ++e) comb[(size_t)t * NE + e] = c[e];
  }
}

// ---------------- fused expert MLP kernel ----------------
// grid = (TT/MT, NE+NS), block = 256 (8 waves). Each block: 32-token tile, one expert.
// Each wave runs TWO 16-row M-subtiles against each B-fragment (2 wmma per weight load).
template <typename WT>
__global__ __launch_bounds__(256)
void moe_expert(const WT* __restrict__ x,
                const WT* __restrict__ re_gu, const WT* __restrict__ re_dn,
                const WT* __restrict__ se_gu, const WT* __restrict__ se_dn,
                const float* __restrict__ comb, float* __restrict__ out) {
  extern __shared__ char smem[];
  unsigned short* xl   = (unsigned short*)smem;                         // MT*DH bf16 = 128 KB
  unsigned short* hl   = (unsigned short*)(smem + MT * DH * 2);         // MT*II bf16 = 64 KB
  float*          scv  = (float*)(smem + MT * DH * 2 + MT * II * 2);    // MT f32
  int*            flag = (int*)(scv + MT);

  const int tid  = threadIdx.x;
  const int lane = tid & 31;
  const int w    = tid >> 5;
  const int t0   = blockIdx.x * MT;
  const int e    = blockIdx.y;

  const WT* W1; const WT* W2;
  if (e < NE) { W1 = re_gu + (size_t)e * (2 * II) * DH; W2 = re_dn + (size_t)e * DH * II; }
  else        { int s = e - NE;
                W1 = se_gu + (size_t)s * (2 * II) * DH; W2 = se_dn + (size_t)s * DH * II; }

  if (tid < MT) scv[tid] = (e < NE) ? comb[(size_t)(t0 + tid) * NE + e] : 0.5f;
  __syncthreads();
  if (e < NE) {
    if (tid == 0) {
      int nz = 0;
      for (int i = 0; i < MT; ++i) nz |= (scv[i] != 0.0f);
      *flag = nz;
    }
    __syncthreads();
    if (!*flag) return;   // no token in this tile routed to expert e
  }

  // stage x tile (tile rows are contiguous in memory -> linear copy)
  if constexpr (sizeof(WT) == 2) {
    // CDNA5 async direct global->LDS copy (no VGPR bounce), tracked by ASYNCcnt
    const char* gsrc = (const char*)(x + (size_t)t0 * DH);
    const unsigned lbase = (unsigned)(uintptr_t)xl;     // LDS byte offset (low 32 bits)
    for (int i = tid; i < MT * DH * 2 / 16; i += 256) {
      unsigned      lofs = lbase + (unsigned)i * 16u;
      const char*   gptr = gsrc + (size_t)i * 16u;
      asm volatile("global_load_async_to_lds_b128 %0, %1, off"
                   :: "v"(lofs), "v"(gptr) : "memory");
    }
    wait_asynccnt0();
  } else {
    const WT* src = x + (size_t)t0 * DH;
    for (int i = tid; i < MT * DH; i += 256) xl[i] = bfbits((float)src[i]);
  }
  __syncthreads();

  const int nlane = lane & 15;
  const int moff  = (lane >> 4) << 3;

  // ---- GEMM1: GU = X @ W1^T, fused SiLU(g)*u -> H (bf16 in LDS) ----
  for (int j = 0; j < 8; ++j) {
    const int gbase = w * 128 + j * 16;                 // gate half: 0..1008
    const WT* rg = W1 + (size_t)(gbase + nlane) * DH;
    const WT* ru = W1 + (size_t)(gbase + II + nlane) * DH;
    __builtin_prefetch(rg, 0, 1);
    __builtin_prefetch(ru, 0, 1);
    v8f accg0 = {}; v8f accg1 = {}; v8f accu0 = {}; v8f accu1 = {};
    for (int kb = 0; kb < DH; kb += 32) {
      v16bf a0 = frag_a_lds(xl,           DH, kb, lane);   // tokens 0-15
      v16bf a1 = frag_a_lds(xl + 16 * DH, DH, kb, lane);   // tokens 16-31
      v16bf bg = frag_b(rg, kb, lane);
      v16bf bu = frag_b(ru, kb, lane);
      accg0 = __builtin_amdgcn_wmma_f32_16x16x32_bf16(false, a0, false, bg, (short)0, accg0, false, false);
      accg1 = __builtin_amdgcn_wmma_f32_16x16x32_bf16(false, a1, false, bg, (short)0, accg1, false, false);
      accu0 = __builtin_amdgcn_wmma_f32_16x16x32_bf16(false, a0, false, bu, (short)0, accu0, false, false);
      accu1 = __builtin_amdgcn_wmma_f32_16x16x32_bf16(false, a1, false, bu, (short)0, accu1, false, false);
    }
    #pragma unroll
    for (int r = 0; r < 8; ++r) {
      float g0 = accg0[r];
      float h0 = g0 * fast_rcp(1.0f + __expf(-g0)) * accu0[r];   // silu(g)*u
      hl[(size_t)(r + moff) * II + gbase + nlane] = bfbits(h0);
      float g1 = accg1[r];
      float h1 = g1 * fast_rcp(1.0f + __expf(-g1)) * accu1[r];
      hl[(size_t)(16 + r + moff) * II + gbase + nlane] = bfbits(h1);
    }
  }
  __syncthreads();

  // ---- GEMM2: OUT += (H @ W2^T) * combine_weight ----
  for (int j = 0; j < 16; ++j) {
    const int nb = w * 256 + j * 16;                     // 0..2032
    const WT* rd = W2 + (size_t)(nb + nlane) * II;
    __builtin_prefetch(rd, 0, 1);
    v8f acc0 = {}; v8f acc1 = {};
    for (int kb = 0; kb < II; kb += 32) {
      v16bf a0 = frag_a_lds(hl,           II, kb, lane);
      v16bf a1 = frag_a_lds(hl + 16 * II, II, kb, lane);
      v16bf b  = frag_b(rd, kb, lane);
      acc0 = __builtin_amdgcn_wmma_f32_16x16x32_bf16(false, a0, false, b, (short)0, acc0, false, false);
      acc1 = __builtin_amdgcn_wmma_f32_16x16x32_bf16(false, a1, false, b, (short)0, acc1, false, false);
    }
    #pragma unroll
    for (int r = 0; r < 8; ++r) {
      float s0 = scv[r + moff];
      if (s0 != 0.0f)
        atomicAdd(out + (size_t)(t0 + r + moff) * DH + nb + nlane, acc0[r] * s0);
      float s1 = scv[16 + r + moff];
      if (s1 != 0.0f)
        atomicAdd(out + (size_t)(t0 + 16 + r + moff) * DH + nb + nlane, acc1[r] * s1);
    }
  }
}

extern "C" void kernel_launch(void* const* d_in, const int* in_sizes, int n_in,
                              void* d_out, int out_size, void* d_ws, size_t ws_size,
                              hipStream_t stream) {
  const float* x     = (const float*)d_in[0];
  const float* gw    = (const float*)d_in[1];
  const float* gb    = (const float*)d_in[2];
  const float* re_gu = (const float*)d_in[3];
  const float* re_dn = (const float*)d_in[4];
  const float* se_gu = (const float*)d_in[5];
  const float* se_dn = (const float*)d_in[6];
  float* out  = (float*)d_out;                 // [TT*DH] out, then [NE] expert_load
  float* comb = (float*)d_ws;                  // [TT*NE] combine weights

  // 1) zero output + expert_load
  moe_zero<<<4096, 256, 0, stream>>>(out, out_size);

  // 2) gating -> comb + expert_load
  moe_gate<<<TT / 8, 256, 0, stream>>>(x, gw, gb, comb, out + (size_t)TT * DH);

  // 3) fused expert MLPs, WMMA bf16
  const size_t smemB  = (size_t)MT * DH * 2 + (size_t)MT * II * 2 + MT * 4 + 64;
  const size_t combB  = (size_t)TT * NE * 4;
  const size_t w1e    = (size_t)2 * II * DH;          // elements per expert (gate_up)
  const size_t w2e    = (size_t)DH * II;              // elements per expert (down)
  const size_t xE     = (size_t)TT * DH;
  const size_t needB  = combB + ((NE + NS) * (w1e + w2e) + xE) * 2;

  if (ws_size >= needB) {
    // pre-convert weights + activations to bf16 once -> GEMM hot loop reads bf16 directly
    unsigned short* w1b = (unsigned short*)((char*)d_ws + combB);
    unsigned short* w2b = w1b + (NE + NS) * w1e;
    unsigned short* xb  = w2b + (NE + NS) * w2e;
    moe_cvt_bf16<<<4096, 256, 0, stream>>>(re_gu, w1b,             (int)(NE * w1e));
    moe_cvt_bf16<<<4096, 256, 0, stream>>>(se_gu, w1b + NE * w1e,  (int)(NS * w1e));
    moe_cvt_bf16<<<4096, 256, 0, stream>>>(re_dn, w2b,             (int)(NE * w2e));
    moe_cvt_bf16<<<4096, 256, 0, stream>>>(se_dn, w2b + NE * w2e,  (int)(NS * w2e));
    moe_cvt_bf16<<<4096, 256, 0, stream>>>(x,     xb,              (int)xE);
    moe_expert<unsigned short><<<dim3(TT / MT, NE + NS), 256, smemB, stream>>>(
        xb, w1b, w2b, w1b + NE * w1e, w2b + NE * w2e, comb, out);
  } else {
    // fallback: convert f32 weights in the GEMM loop
    moe_expert<float><<<dim3(TT / MT, NE + NS), 256, smemB, stream>>>(
        x, re_gu, re_dn, se_gu, se_dn, comb, out);
  }
}
